// Seq2SeqAttentionDecoder_27384711479550
// MI455X (gfx1250) — matrix operand order repaired and checked by reading, not compiled, not run on hardware
//
#include <hip/hip_runtime.h>
#include <cstdint>

#define B_     32
#define T_     64
#define L_     128
#define H_     1024
#define EMB_   512
#define VOCAB_ 32000

typedef __attribute__((ext_vector_type(16))) __bf16        v16bf;
typedef __attribute__((ext_vector_type(8)))  float         v8f;
typedef __attribute__((ext_vector_type(4)))  unsigned int  u32x4;
typedef __attribute__((ext_vector_type(8)))  int           i32x8;
typedef __attribute__((ext_vector_type(4)))  int           i32x4;

__device__ __forceinline__ float sigmoidf_(float x){ return 1.0f / (1.0f + __expf(-x)); }

struct U4x2 { uint4 lo, hi; };
__device__ __forceinline__ v16bf cast16(uint4 lo, uint4 hi){
  U4x2 p{lo, hi};
  return __builtin_bit_cast(v16bf, p);
}

// ---- WMMA fragment loaders (CDNA5 ISA 7.12.2, wave32) ----
// A 16x32 bf16 MxK: lane m = lane&15, half = lane>>4; per-lane data = cols
// [half*16, half*16+16) of row m  => two b128 loads.
__device__ __forceinline__ v16bf load_a16(const __bf16* __restrict__ A, int ld,
                                          int mbase, int kk, int lane){
  const int m = lane & 15, half = lane >> 4;
  const __bf16* p = A + (size_t)(mbase + m) * ld + kk + half * 16;
  return cast16(*(const uint4*)p, *(const uint4*)(p + 8));
}

// B 32x16 bf16 KxN from row-major W[out][k]: lane n = lane&15, koff = (lane>>4)*16;
// per-lane data = 16 consecutive bf16 of row (obase+n) at kbase+koff => two b128 loads.
__device__ __forceinline__ v16bf load_b16(const __bf16* __restrict__ W, int ldw,
                                          int obase, int kbase, int lane){
  const int n = lane & 15, koff = (lane >> 4) * 16;
  const __bf16* p = W + (size_t)(obase + n) * ldw + kbase + koff;
  return cast16(*(const uint4*)p, *(const uint4*)(p + 8));
}

__device__ __forceinline__ v8f wmma_bf16(v16bf a, v16bf b, v8f c){
  return __builtin_amdgcn_wmma_f32_16x16x32_bf16(false, a, false, b, (short)0, c, false, false);
}

// ---- TDM: DMA a 2D bf16 tile (32 elems x 64 rows, row stride 1024 elems) into LDS ----
// D# built per CDNA5 ISA (group0 128b, group1 256b, groups2/3 zero).
// This toolchain exposes the 6-arg builtin: (u32x4, i32x8, i32x4, i32x4, i32x8, i32 cpol).
__device__ __forceinline__ void tdm_load_b_tile(const __bf16* gptr, unsigned lds_byte_off){
  const unsigned long long ga = (unsigned long long)(uintptr_t)gptr;
  u32x4 g0;
  g0[0] = 1u;                                       // count=1, user descriptor
  g0[1] = lds_byte_off;                             // lds_addr [63:32]
  g0[2] = (unsigned)(ga & 0xFFFFFFFFu);             // global_addr [95:64]
  g0[3] = (unsigned)((ga >> 32) & 0x1FFFFFFu)       // global_addr [120:96]
        | (2u << 30);                               // type=2 ("image")
  const unsigned TD = 1u << 20;                     // huge tensor dims: never OOB
  i32x8 g1;
  g1[0] = (int)(1u << 16);                          // data_size=1 (2 bytes); mask/flags=0
  g1[1] = (int)((TD & 0xFFFFu) << 16);              // tensor_dim0[15:0] at [63:48]
  g1[2] = (int)(((TD >> 16) & 0xFFFFu)              // tensor_dim0[31:16]
        | ((TD & 0xFFFFu) << 16));                  // tensor_dim1[15:0]
  g1[3] = (int)(((TD >> 16) & 0xFFFFu)              // tensor_dim1[31:16]
        | (32u << 16));                             // tile_dim0 = 32 elements (64B row)
  g1[4] = (int)(64u);                               // tile_dim1 = 64 rows; tile_dim2 = 0
  g1[5] = (int)((unsigned)H_);                      // tensor_dim0_stride[31:0] = 1024 elems
  g1[6] = 0;                                        // stride hi / dim1_stride lo
  g1[7] = 0;
  i32x4 gz4 = {0, 0, 0, 0};
  i32x8 gz8 = {0, 0, 0, 0, 0, 0, 0, 0};
  __builtin_amdgcn_tensor_load_to_lds(g0, g1, gz4, gz4, gz8, 0);
}

// ---- Prep kernels ----

__global__ void f32_to_bf16_kernel(const float* __restrict__ src, __bf16* __restrict__ dst, size_t n){
  size_t i = (size_t)blockIdx.x * blockDim.x + threadIdx.x;
  const size_t stride = (size_t)gridDim.x * blockDim.x;
  for (; i < n; i += stride) dst[i] = (__bf16)src[i];
}

__global__ void embed_kernel(const int* __restrict__ X, const float* __restrict__ Wemb,
                             __bf16* __restrict__ Xemb){
  const size_t n = (size_t)B_ * T_ * EMB_;
  size_t i = (size_t)blockIdx.x * blockDim.x + threadIdx.x;
  const size_t stride = (size_t)gridDim.x * blockDim.x;
  for (; i < n; i += stride){
    const size_t bt = i / EMB_;
    const int k = (int)(i % EMB_);
    Xemb[i] = (__bf16)Wemb[(size_t)X[bt] * EMB_ + k];
  }
}

// ---- Attention (per step): scores -> masked softmax -> ctx; assembles contiguous
// inp0[b] = [emb(x_t) | ctx] in bf16 so GRU0's A operand is a single row-major matrix.

__global__ void attn_kernel(const float* __restrict__ enc, const float* __restrict__ hs,
                            const float* __restrict__ H1, const int* __restrict__ valid,
                            const __bf16* __restrict__ Xemb, __bf16* __restrict__ inp0b, int t){
  __shared__ float red[256];
  __shared__ float sc[L_];
  __shared__ float stat;
  const int b = blockIdx.x, tid = threadIdx.x;
  const float* q = (t == 0) ? (hs + (size_t)1 * B_ * H_ + (size_t)b * H_)
                            : (H1 + (size_t)(t - 1) * B_ * H_ + (size_t)b * H_);
  const float* encb = enc + (size_t)b * L_ * H_;

  // copy this step's embedding slice into inp0 row
  for (int k = tid; k < EMB_; k += 256)
    inp0b[(size_t)b * (EMB_ + H_) + k] = Xemb[((size_t)b * T_ + t) * EMB_ + k];

  const int l = tid & (L_ - 1);
  const int part = tid >> 7;
  {
    const float4* qv = (const float4*)(q + part * 512);
    const float4* ev = (const float4*)(encb + (size_t)l * H_ + part * 512);
    float s = 0.f;
    for (int k = 0; k < 128; ++k){
      const float4 a = qv[k], e = ev[k];
      s += a.x * e.x + a.y * e.y + a.z * e.z + a.w * e.w;
    }
    red[tid] = s;
  }
  __syncthreads();
  if (part == 0){
    float s = (red[tid] + red[tid + 128]) * 0.03125f;  // 1/sqrt(1024)
    if (l >= valid[b]) s = -1e6f;
    sc[l] = s;
  }
  __syncthreads();
  if (tid == 0){
    float mx = sc[0];
    for (int i = 1; i < L_; ++i) mx = fmaxf(mx, sc[i]);
    stat = mx;
  }
  __syncthreads();
  if (tid < L_) sc[tid] = __expf(sc[tid] - stat);
  __syncthreads();
  if (tid == 0){
    float sum = 0.f;
    for (int i = 0; i < L_; ++i) sum += sc[i];
    stat = 1.0f / sum;
  }
  __syncthreads();
  const float inv = stat;
  for (int hh = tid; hh < H_; hh += 256){
    float acc = 0.f;
    for (int ll = 0; ll < L_; ++ll) acc += sc[ll] * encb[(size_t)ll * H_ + hh];
    inp0b[(size_t)b * (EMB_ + H_) + EMB_ + hh] = (__bf16)(acc * inv);
  }
}

// ---- GRU cell at h=0 (2 waves/block, 16 output cols/block, 3 gate accs share A) ----

__global__ __launch_bounds__(64)
void gru0_kernel(const __bf16* __restrict__ Wb, const float* __restrict__ b_ih,
                 const float* __restrict__ b_hh, const __bf16* __restrict__ inp0b,
                 float* __restrict__ h0f, __bf16* __restrict__ h0b){
  const int tid = threadIdx.x, lane = tid & 31, mbase = (tid >> 5) * 16;
  const int jbase = blockIdx.x * 16;
  const v8f zero = {0,0,0,0,0,0,0,0};
  v8f ar = zero, az = zero, an = zero;
  for (int kk = 0; kk < EMB_ + H_; kk += 32){
    const v16bf a = load_a16(inp0b, EMB_ + H_, mbase, kk, lane);
    ar = wmma_bf16(a, load_b16(Wb, EMB_ + H_, jbase,        kk, lane), ar);
    az = wmma_bf16(a, load_b16(Wb, EMB_ + H_, jbase + H_,   kk, lane), az);
    an = wmma_bf16(a, load_b16(Wb, EMB_ + H_, jbase + 2*H_, kk, lane), an);
  }
  const int n = lane & 15, half = lane >> 4;
  const int o = jbase + n;
#pragma unroll
  for (int r = 0; r < 8; ++r){
    const int m = mbase + r + 8 * half;
    const float rr = sigmoidf_(ar[r] + b_ih[o]        + b_hh[o]);
    const float zz = sigmoidf_(az[r] + b_ih[o + H_]   + b_hh[o + H_]);
    const float nn = tanhf    (an[r] + b_ih[o + 2*H_] + rr * b_hh[o + 2*H_]);
    const float h = (1.0f - zz) * nn;
    h0f[(size_t)m * H_ + o] = h;
    h0b[(size_t)m * H_ + o] = (__bf16)h;
  }
}

__global__ __launch_bounds__(64)
void gru1_kernel(const __bf16* __restrict__ Wb, const float* __restrict__ b_ih,
                 const float* __restrict__ b_hh, const __bf16* __restrict__ h0b,
                 float* __restrict__ h1f, __bf16* __restrict__ h1b){
  const int tid = threadIdx.x, lane = tid & 31, mbase = (tid >> 5) * 16;
  const int jbase = blockIdx.x * 16;
  const v8f zero = {0,0,0,0,0,0,0,0};
  v8f ar = zero, az = zero, an = zero;
  for (int kk = 0; kk < H_; kk += 32){
    const v16bf a = load_a16(h0b, H_, mbase, kk, lane);
    ar = wmma_bf16(a, load_b16(Wb, H_, jbase,        kk, lane), ar);
    az = wmma_bf16(a, load_b16(Wb, H_, jbase + H_,   kk, lane), az);
    an = wmma_bf16(a, load_b16(Wb, H_, jbase + 2*H_, kk, lane), an);
  }
  const int n = lane & 15, half = lane >> 4;
  const int o = jbase + n;
#pragma unroll
  for (int r = 0; r < 8; ++r){
    const int m = mbase + r + 8 * half;
    const float rr = sigmoidf_(ar[r] + b_ih[o]        + b_hh[o]);
    const float zz = sigmoidf_(az[r] + b_ih[o + H_]   + b_hh[o + H_]);
    const float nn = tanhf    (an[r] + b_ih[o + 2*H_] + rr * b_hh[o + 2*H_]);
    const float h = (1.0f - zz) * nn;
    h1f[(size_t)m * H_ + o] = h;
    h1b[(size_t)m * H_ + o] = (__bf16)h;
  }
}

// ---- Phase 2: logits = H1[2048,1024] @ W_ffn^T + b_ffn ----
// Block = 8 waves, block tile M=128 x N=64. W_ffn tiles (64 rows x 32 cols bf16 = 4KB)
// are TDM-DMA'd into LDS (double-buffered, TENSORcnt) and shared by all 8 waves;
// each wave owns one 16-row M tile and 4 WMMA accumulators.

__global__ __launch_bounds__(256)
void logits_kernel(const __bf16* __restrict__ H1b, const __bf16* __restrict__ Wb,
                   const float* __restrict__ b_ffn, float* __restrict__ out){
  __shared__ __align__(16) unsigned short btile[2][64 * 32];   // 2 x 4KB
  const int tid = threadIdx.x, lane = tid & 31, wave = tid >> 5;
  const int mbase = (blockIdx.y * 8 + wave) * 16;
  const int nbase = blockIdx.x * 64;
  const unsigned lds0 = (unsigned)(uintptr_t)(void*)&btile[0][0];
  const unsigned lds1 = (unsigned)(uintptr_t)(void*)&btile[1][0];
  const v8f zero = {0,0,0,0,0,0,0,0};
  v8f acc[4];
#pragma unroll
  for (int j = 0; j < 4; ++j) acc[j] = zero;
  const int n = lane & 15, koff = (lane >> 4) * 16, half = lane >> 4;
  const __bf16* gB = Wb + (size_t)nbase * H_;

  if (wave == 0) tdm_load_b_tile(gB, lds0);

  for (int i = 0; i < H_ / 32; ++i){
    const int kk = i * 32;
    const int cur = i & 1;
    if (wave == 0){
      if (i + 1 < H_ / 32){
        tdm_load_b_tile(gB + (kk + 32), cur ? lds0 : lds1);
        __builtin_amdgcn_s_wait_tensorcnt(1);   // oldest (current buffer) done
      } else {
        __builtin_amdgcn_s_wait_tensorcnt(0);
      }
    }
    __syncthreads();                            // publish LDS tile to all waves

    const v16bf a = load_a16(H1b, H_, mbase, kk, lane);
    if (kk + 32 < H_)
      __builtin_prefetch(&H1b[(size_t)(mbase + (lane & 15)) * H_ + kk + 32], 0, 3);
#pragma unroll
    for (int j = 0; j < 4; ++j){
      const unsigned short* rp = &btile[cur][(unsigned)((j * 16 + n) * 32 + koff)];
      const v16bf bfrag = cast16(*(const uint4*)rp, *(const uint4*)(rp + 8));
      acc[j] = wmma_bf16(a, bfrag, acc[j]);
    }
    __syncthreads();                            // buffer free before TDM reuses it
  }

#pragma unroll
  for (int j = 0; j < 4; ++j){
    const int v = nbase + j * 16 + n;
    const float bias = b_ffn[v];
#pragma unroll
    for (int r = 0; r < 8; ++r){
      const int m = mbase + r + 8 * half;       // GEMM row = t*B + b
      const int tt = m >> 5, bb = m & 31;
      out[((size_t)bb * T_ + tt) * VOCAB_ + v] = acc[j][r] + bias;
    }
  }
}

__global__ void final_hidden_kernel(const float* __restrict__ h0f, const float* __restrict__ H1,
                                    float* __restrict__ outF){
  const int i = blockIdx.x * blockDim.x + threadIdx.x;
  const int n = B_ * H_;
  if (i < n) outF[i] = h0f[i];
  else if (i < 2 * n) outF[i] = H1[(size_t)(T_ - 1) * B_ * H_ + (i - n)];
}

extern "C" void kernel_launch(void* const* d_in, const int* in_sizes, int n_in,
                              void* d_out, int out_size, void* d_ws, size_t ws_size,
                              hipStream_t stream){
  (void)in_sizes; (void)n_in; (void)out_size; (void)ws_size;
  const int*   X    = (const int*)  d_in[0];
  const float* enc  = (const float*)d_in[1];
  const float* hs   = (const float*)d_in[2];
  const int*   vlen = (const int*)  d_in[3];
  const float* Wemb = (const float*)d_in[4];
  const float* Wih0 = (const float*)d_in[5];
  /* d_in[6] = W_hh0 : unused (GRU evaluated at h=0) */
  const float* bih0 = (const float*)d_in[7];
  const float* bhh0 = (const float*)d_in[8];
  const float* Wih1 = (const float*)d_in[9];
  /* d_in[10] = W_hh1 : unused */
  const float* bih1 = (const float*)d_in[11];
  const float* bhh1 = (const float*)d_in[12];
  const float* Wffn = (const float*)d_in[13];
  const float* bffn = (const float*)d_in[14];
  float* out = (float*)d_out;

  char* ws = (char*)d_ws;
  size_t off = 0;
  auto alloc = [&](size_t bytes) -> char* {
    char* p = ws + off;
    off = (off + bytes + 255) & ~(size_t)255;
    return p;
  };
  __bf16* Wih0b = (__bf16*)alloc((size_t)3 * H_ * (EMB_ + H_) * 2);
  __bf16* Wih1b = (__bf16*)alloc((size_t)3 * H_ * H_ * 2);
  __bf16* Wffnb = (__bf16*)alloc((size_t)VOCAB_ * H_ * 2);
  __bf16* Xemb  = (__bf16*)alloc((size_t)B_ * T_ * EMB_ * 2);
  float*  H1    = (float*) alloc((size_t)T_ * B_ * H_ * 4);
  __bf16* H1b   = (__bf16*)alloc((size_t)T_ * B_ * H_ * 2);
  float*  h0f   = (float*) alloc((size_t)B_ * H_ * 4);
  __bf16* h0b   = (__bf16*)alloc((size_t)B_ * H_ * 2);
  __bf16* inp0b = (__bf16*)alloc((size_t)B_ * (EMB_ + H_) * 2);

  // One-time prep: bf16 weight copies + full embedding gather.
  f32_to_bf16_kernel<<<2048, 256, 0, stream>>>(Wih0, Wih0b, (size_t)3 * H_ * (EMB_ + H_));
  f32_to_bf16_kernel<<<2048, 256, 0, stream>>>(Wih1, Wih1b, (size_t)3 * H_ * H_);
  f32_to_bf16_kernel<<<8192, 256, 0, stream>>>(Wffn, Wffnb, (size_t)VOCAB_ * H_);
  embed_kernel<<<2048, 256, 0, stream>>>(X, Wemb, Xemb);

  // Sequential recurrence (attention -> GRU0 -> GRU1), storing all h1[t].
  for (int t = 0; t < T_; ++t){
    attn_kernel<<<B_, 256, 0, stream>>>(enc, hs, H1, vlen, Xemb, inp0b, t);
    gru0_kernel<<<H_ / 16, 64, 0, stream>>>(Wih0b, bih0, bhh0, inp0b, h0f, h0b);
    gru1_kernel<<<H_ / 16, 64, 0, stream>>>(Wih1b, bih1, bhh1, h0b,
                                            H1 + (size_t)t * B_ * H_,
                                            H1b + (size_t)t * B_ * H_);
  }

  // One big batched logits GEMM: [T*B=2048, 1024] x [1024, 32000], TDM-fed.
  logits_kernel<<<dim3(VOCAB_ / 64, (T_ * B_) / (8 * 16)), 256, 0, stream>>>(H1b, Wffnb, bffn, out);
  final_hidden_kernel<<<(2 * B_ * H_ + 255) / 256, 256, 0, stream>>>(h0f, H1, out + (size_t)B_ * T_ * VOCAB_);
}
